// LocalPointDecoder_4853313044948
// MI455X (gfx1250) — matrix operand order repaired and checked by reading, not compiled, not run on hardware
//
#include <hip/hip_runtime.h>
#include <hip/hip_bf16.h>

typedef _Float16 half8  __attribute__((ext_vector_type(8)));
typedef _Float16 half16 __attribute__((ext_vector_type(16)));
typedef float    float8 __attribute__((ext_vector_type(8)));
typedef unsigned int uint4v __attribute__((ext_vector_type(4)));
typedef int          int8v  __attribute__((ext_vector_type(8)));
typedef int          int4v  __attribute__((ext_vector_type(4)));

union H16 { half16 v; half8 h8[2]; _Float16 h[16]; };

#define NQ   16384
#define MP   4096
#define CF   128
#define HD   256
#define NBLK 5

#if __has_include(<hip/amd_detail/amd_gfx1250_TDM.h>)
#define TDM_6ARG 1
#else
#define TDM_6ARG 0
#endif

static __device__ __forceinline__ float8 wmma16(const H16& a, const H16& b, float8 c) {
  // D = A(16x32 f16) * B(32x16 f16) + C(16x16 f32)
  return __builtin_amdgcn_wmma_f32_16x16x32_f16(false, a.v, false, b.v,
                                                (short)0, c, false, false);
}

// Tensor Data Mover: DMA a [tile_dim1=128 rows][tile_dim0=32 elems] f16 tile
// (row stride 4096 elems) from global into LDS, rows packed 64B apart.
static __device__ __forceinline__ void tdm_load_tile(const _Float16* gsrc,
                                                     unsigned ldsOff) {
  const unsigned long long ga = (unsigned long long)(size_t)gsrc;
  uint4v g0;
  g0[0] = 1u;                                   // count=1, no gather
  g0[1] = ldsOff;                               // lds_addr (bytes)
  g0[2] = (unsigned)ga;                         // global_addr[31:0]
  g0[3] = (unsigned)(ga >> 32) | (2u << 30);    // global_addr[56:32] | type=2
  int8v g1;
  g1[0] = (int)(1u << 16);                      // data_size = 2 bytes
  g1[1] = (int)(4096u << 16);                   // tensor_dim0[15:0] @ bits 63:48
  g1[2] = (int)(128u << 16);                    // tensor_dim1[15:0] @ bits 95:80
  g1[3] = (int)(32u << 16);                     // tile_dim0 = 32   @ bits 127:112
  g1[4] = 128;                                  // tile_dim1 = 128, tile_dim2 = 0
  g1[5] = 4096;                                 // tensor_dim0_stride[31:0]
  g1[6] = 0;                                    // stride hi / dim1_stride = 0
  g1[7] = 0;
  const int4v z4 = (int4v)0;
#if TDM_6ARG
  const int8v z8 = (int8v)0;
  __builtin_amdgcn_tensor_load_to_lds(g0, g1, z4, z4, z8, 0);
#else
  __builtin_amdgcn_tensor_load_to_lds(g0, g1, z4, z4, 0);
#endif
}

// ---------------------------------------------------------------------------
// Prep: fea [M][C] f32 -> feaT [C][M] f16  (B-fragment friendly: contiguous K)
// ---------------------------------------------------------------------------
__global__ __launch_bounds__(256) void k_feaT(const float* __restrict__ fea,
                                              _Float16* __restrict__ feaT) {
  int i = blockIdx.x * 256 + threadIdx.x;          // over CF*MP
  if (i < CF * MP) {
    int c = i >> 12;            // / MP
    int m = i & (MP - 1);
    feaT[i] = (_Float16)fea[m * CF + c];
  }
}

// Prep: straight f32 -> f16 convert (weights keep [out][in] row-major layout)
__global__ __launch_bounds__(256) void k_cvt(const float* __restrict__ src,
                                             _Float16* __restrict__ dst, int n) {
  int i = blockIdx.x * 256 + threadIdx.x;
  if (i < n) dst[i] = (_Float16)src[i];
}

// ---------------------------------------------------------------------------
// Fused sample_point_feature: softmax(-(dist+eps)^2/var) @ fea -> cfeat f16.
// One wave = one 16-query tile; 4 waves share TDM-staged fea tiles in LDS,
// double-buffered and overlapped with WMMA via TENSORcnt.
// ---------------------------------------------------------------------------
__global__ __launch_bounds__(128) void k_sample(const float* __restrict__ p,
                                                const float* __restrict__ pp,
                                                const _Float16* __restrict__ feaT,
                                                _Float16* __restrict__ cfeat) {
  __shared__ __align__(16) _Float16 feaBuf[2][CF * 32];  // 2 x 8KB tiles
  __shared__ float ldsPP[4][96];    // per-wave 32 pts * xyz
  __shared__ float ldsSum[4][16];   // per-wave row sums
  const int lane = threadIdx.x & 31;
  const int wave = threadIdx.x >> 5;
  const int row  = lane & 15;       // query row this lane feeds in A
  const int hi   = lane >> 4;       // half-wave select
  const int koff = hi * 8;          // A-layout K offset (0..7 | 8..15)
  const int n0   = (blockIdx.x * 4 + wave) * 16;

  const float qx = p[(n0 + row) * 3 + 0];
  const float qy = p[(n0 + row) * 3 + 1];
  const float qz = p[(n0 + row) * 3 + 2];

  float8 acc[8];
#pragma unroll
  for (int g = 0; g < 8; ++g) acc[g] = (float8)0.f;
  float sumexp = 0.f;

  const unsigned buf0 = (unsigned)(size_t)(void*)&feaBuf[0][0];
  const unsigned buf1 = (unsigned)(size_t)(void*)&feaBuf[1][0];

  // prologue: DMA tile 0
  if (wave == 0) tdm_load_tile(feaT, buf0);

  const int NT = MP / 32;           // 128 tiles
  for (int t = 0; t < NT; ++t) {
    const int cur = t & 1;
    // issue next tile into the other buffer; wait until tile t landed
    if (wave == 0) {
      if (t + 1 < NT) {
        tdm_load_tile(feaT + (t + 1) * 32, cur ? buf0 : buf1);
        __builtin_amdgcn_s_wait_tensorcnt((short)1);
      } else {
        __builtin_amdgcn_s_wait_tensorcnt((short)0);
      }
    }

    // stage pp tile (per-wave slice) and build exp-weight A fragment in ISA
    // layout: h[0..7] -> K = koff+0..7 ; h[8..15] -> K = 16+koff+0..7
    const float* src = pp + t * 96;
    ldsPP[wave][lane * 3 + 0] = src[lane * 3 + 0];
    ldsPP[wave][lane * 3 + 1] = src[lane * 3 + 1];
    ldsPP[wave][lane * 3 + 2] = src[lane * 3 + 2];

    H16 a;
#pragma unroll
    for (int j = 0; j < 16; ++j) {
      const int K = (j < 8) ? (koff + j) : (16 + koff + (j - 8));
      const float px = ldsPP[wave][K * 3 + 0];
      const float py = ldsPP[wave][K * 3 + 1];
      const float pz = ldsPP[wave][K * 3 + 2];
      const float dx = px - qx, dy = py - qy, dz = pz - qz;
      const float dist = sqrtf(dx * dx + dy * dy + dz * dz);
      const float tt = dist + 1e-5f;
      const float w = __expf(-tt * tt * 100.0f);   // 1/VAR = 100, logit <= 0
      sumexp += w;
      a.h[j] = (_Float16)w;
    }

    __syncthreads();   // tile t visible to all waves

    // 8 column groups of fea (C = 128) from the LDS tile
#pragma unroll
    for (int g = 0; g < 8; ++g) {
      const int c = g * 16 + row;
      const _Float16* bp = &feaBuf[cur][c * 32 + hi * 8];
      H16 b;
      b.h8[0] = *(const half8*)(bp);
      b.h8[1] = *(const half8*)(bp + 16);
      acc[g] = wmma16(a, b, acc[g]);
    }

    __syncthreads();   // all waves done with buf[cur] before it is re-DMA'd
  }

  // row sum: pair (lane, lane^16) covers all 32 K of a row
  sumexp += __shfl_xor(sumexp, 16, 32);
  if (hi == 0) ldsSum[wave][row] = sumexp;
  float rinv[8];
#pragma unroll
  for (int r = 0; r < 8; ++r) rinv[r] = 1.0f / ldsSum[wave][hi * 8 + r];

  // C layout: acc[g] elem r -> (row hi*8+r, col g*16+(lane&15))
#pragma unroll
  for (int g = 0; g < 8; ++g) {
    const int c = g * 16 + row;
#pragma unroll
    for (int r = 0; r < 8; ++r)
      cfeat[(size_t)(n0 + hi * 8 + r) * CF + c] = (_Float16)(acc[g][r] * rinv[r]);
  }
}

// ---------------------------------------------------------------------------
// MLP stack: net = p@Wp + bp; 5x { net += c@Wc+bc; resnet(fc1(relu(fc0(relu)))) }
// One wave = 16 rows; net kept as 16 persistent f32 WMMA accumulators.
// ---------------------------------------------------------------------------
__global__ __launch_bounds__(64) void k_mlp(const float* __restrict__ p,
                                            const _Float16* __restrict__ cfeat,
                                            const float* __restrict__ fc_p_W,
                                            const float* __restrict__ fc_p_b,
                                            const _Float16* __restrict__ Wc,
                                            const float* __restrict__ bc,
                                            const _Float16* __restrict__ W0,
                                            const float* __restrict__ b0,
                                            const _Float16* __restrict__ W1,
                                            const float* __restrict__ b1,
                                            const float* __restrict__ out_W,
                                            const float* __restrict__ out_b,
                                            float* __restrict__ out) {
  __shared__ _Float16 act[2][16 * HD];   // relu(net) f16, per wave
  __shared__ _Float16 acth[2][16 * HD];  // relu(h)   f16, per wave
  __shared__ float qpt[2][16][3];
  const int lane  = threadIdx.x & 31;
  const int wave  = threadIdx.x >> 5;
  const int col16 = lane & 15;
  const int hi    = lane >> 4;
  const int koff  = hi * 8;
  const int n0    = (blockIdx.x * 2 + wave) * 16;

  if (lane < 16) {
    qpt[wave][lane][0] = p[(n0 + lane) * 3 + 0];
    qpt[wave][lane][1] = p[(n0 + lane) * 3 + 1];
    qpt[wave][lane][2] = p[(n0 + lane) * 3 + 2];
  }

  float8 net[16];
#pragma unroll
  for (int g = 0; g < 16; ++g) {
    const int h = g * 16 + col16;
    const float w0 = fc_p_W[h * 3 + 0];
    const float w1 = fc_p_W[h * 3 + 1];
    const float w2 = fc_p_W[h * 3 + 2];
    const float bb = fc_p_b[h];
#pragma unroll
    for (int r = 0; r < 8; ++r) {
      const int M = hi * 8 + r;
      net[g][r] = bb + qpt[wave][M][0] * w0 + qpt[wave][M][1] * w1
                     + qpt[wave][M][2] * w2;
    }
  }

  for (int i = 0; i < NBLK; ++i) {
    const _Float16* Wci = Wc + (size_t)i * HD * CF;
    const _Float16* W0i = W0 + (size_t)i * HD * HD;
    const _Float16* W1i = W1 + (size_t)i * HD * HD;

    // net += cfeat @ Wc^T + bc ; stash relu(net) as f16
#pragma unroll
    for (int g = 0; g < 16; ++g) {
      const int h = g * 16 + col16;
      for (int kt = 0; kt < 4; ++kt) {        // K = 128
        H16 a, b;
        const _Float16* ap = cfeat + (size_t)(n0 + col16) * CF + kt * 32;
        a.h8[0] = *(const half8*)(ap + koff);
        a.h8[1] = *(const half8*)(ap + 16 + koff);
        const _Float16* bp = Wci + (size_t)h * CF + kt * 32 + hi * 8;
        b.h8[0] = *(const half8*)(bp);
        b.h8[1] = *(const half8*)(bp + 16);
        net[g] = wmma16(a, b, net[g]);
      }
      const float bias = bc[i * HD + h];
#pragma unroll
      for (int r = 0; r < 8; ++r) {
        net[g][r] += bias;
        act[wave][(hi * 8 + r) * HD + h] = (_Float16)fmaxf(net[g][r], 0.f);
      }
    }

    // h = relu(net) @ W0^T + b0 ; stash relu(h)
#pragma unroll
    for (int g = 0; g < 16; ++g) {
      const int gg = g * 16 + col16;
      float8 hacc = (float8)0.f;
      for (int kt = 0; kt < 8; ++kt) {        // K = 256
        H16 a, b;
        const _Float16* ap = &act[wave][col16 * HD + kt * 32];
        a.h8[0] = *(const half8*)(ap + koff);
        a.h8[1] = *(const half8*)(ap + 16 + koff);
        const _Float16* bp = W0i + (size_t)gg * HD + kt * 32 + hi * 8;
        b.h8[0] = *(const half8*)(bp);
        b.h8[1] = *(const half8*)(bp + 16);
        hacc = wmma16(a, b, hacc);
      }
      const float bias = b0[i * HD + gg];
#pragma unroll
      for (int r = 0; r < 8; ++r)
        acth[wave][(hi * 8 + r) * HD + gg] = (_Float16)fmaxf(hacc[r] + bias, 0.f);
    }

    // net += relu(h) @ W1^T + b1
#pragma unroll
    for (int g = 0; g < 16; ++g) {
      const int h = g * 16 + col16;
      for (int kt = 0; kt < 8; ++kt) {        // K = 256
        H16 a, b;
        const _Float16* ap = &acth[wave][col16 * HD + kt * 32];
        a.h8[0] = *(const half8*)(ap + koff);
        a.h8[1] = *(const half8*)(ap + 16 + koff);
        const _Float16* bp = W1i + (size_t)h * HD + kt * 32 + hi * 8;
        b.h8[0] = *(const half8*)(bp);
        b.h8[1] = *(const half8*)(bp + 16);
        net[g] = wmma16(a, b, net[g]);
      }
      const float bias = b1[i * HD + h];
#pragma unroll
      for (int r = 0; r < 8; ++r) net[g][r] += bias;
    }
  }

  // out[n] = relu(net) . out_W + out_b  (reduce 16 lanes per half-wave)
  float part[8];
#pragma unroll
  for (int r = 0; r < 8; ++r) part[r] = 0.f;
#pragma unroll
  for (int g = 0; g < 16; ++g) {
    const int h = g * 16 + col16;
    const float w = out_W[h];
#pragma unroll
    for (int r = 0; r < 8; ++r) part[r] += fmaxf(net[g][r], 0.f) * w;
  }
  for (int m = 1; m < 16; m <<= 1) {
#pragma unroll
    for (int r = 0; r < 8; ++r) part[r] += __shfl_xor(part[r], m, 32);
  }
  if (col16 == 0) {
    const float ob = out_b[0];
#pragma unroll
    for (int r = 0; r < 8; ++r) out[n0 + hi * 8 + r] = part[r] + ob;
  }
}

// ---------------------------------------------------------------------------
extern "C" void kernel_launch(void* const* d_in, const int* in_sizes, int n_in,
                              void* d_out, int out_size, void* d_ws, size_t ws_size,
                              hipStream_t stream) {
  const float* p      = (const float*)d_in[0];
  const float* pp     = (const float*)d_in[1];
  const float* fea    = (const float*)d_in[2];
  const float* fc_p_W = (const float*)d_in[3];
  const float* fc_p_b = (const float*)d_in[4];
  const float* fc_c_W = (const float*)d_in[5];
  const float* fc_c_b = (const float*)d_in[6];
  const float* blk0_W = (const float*)d_in[7];
  const float* blk0_b = (const float*)d_in[8];
  const float* blk1_W = (const float*)d_in[9];
  const float* blk1_b = (const float*)d_in[10];
  const float* out_W  = (const float*)d_in[11];
  const float* out_b  = (const float*)d_in[12];
  float* out = (float*)d_out;

  // workspace carve (f16): feaT 1MB | Wc 320KB | W0 640KB | W1 640KB | cfeat 4MB
  char* ws = (char*)d_ws;
  _Float16* feaT  = (_Float16*)(ws + 0);
  _Float16* WcH   = (_Float16*)(ws + 1048576);
  _Float16* W0H   = (_Float16*)(ws + 1376256);
  _Float16* W1H   = (_Float16*)(ws + 2031616);
  _Float16* cfeat = (_Float16*)(ws + 2686976);

  k_feaT<<<(CF * MP + 255) / 256, 256, 0, stream>>>(fea, feaT);
  k_cvt<<<(NBLK * HD * CF + 255) / 256, 256, 0, stream>>>(fc_c_W, WcH, NBLK * HD * CF);
  k_cvt<<<(NBLK * HD * HD + 255) / 256, 256, 0, stream>>>(blk0_W, W0H, NBLK * HD * HD);
  k_cvt<<<(NBLK * HD * HD + 255) / 256, 256, 0, stream>>>(blk1_W, W1H, NBLK * HD * HD);

  k_sample<<<NQ / 64, 128, 0, stream>>>(p, pp, feaT, cfeat);

  k_mlp<<<NQ / 32, 64, 0, stream>>>(p, cfeat, fc_p_W, fc_p_b,
                                    WcH, fc_c_b, W0H, blk0_b, W1H, blk1_b,
                                    out_W, out_b, out);
}
